// DenseBasicBlock_69724499084084
// MI455X (gfx1250) — matrix operand order repaired and checked by reading, not compiled, not run on hardware
//
#include <hip/hip_runtime.h>
#include <hip/hip_bf16.h>
#include <math.h>

typedef _Float16 v16h __attribute__((ext_vector_type(16)));
typedef _Float16 v8h  __attribute__((ext_vector_type(8)));
typedef _Float16 v4h  __attribute__((ext_vector_type(4)));
typedef float    v8f  __attribute__((ext_vector_type(8)));

// Problem constants (from reference)
#define BATCH 128
#define CIN   256
#define HH    32
#define WW    32
#define GOUT  12        // growthRate (real output channels)
#define KRED  (CIN*9)   // 2304 reduction length
#define KT    (KRED/32) // 72 K-tiles of 32
#define COUT_TOTAL (CIN + GOUT) // 268
#define HP    34        // halo'd height (rows -1..32)
#define WP    34        // halo'd width

// ---------------------------------------------------------------------------
// 1) max(|tanh(w)|) over all 12*256*3*3 weights (single block reduction)
// ---------------------------------------------------------------------------
__global__ void absmax_kernel(const float* __restrict__ w, float* __restrict__ omax, int n) {
    __shared__ float red[256];
    float m = 0.0f;
    for (int i = threadIdx.x; i < n; i += 256)
        m = fmaxf(m, fabsf(tanhf(w[i])));
    red[threadIdx.x] = m;
    __syncthreads();
    for (int s = 128; s > 0; s >>= 1) {
        if (threadIdx.x < s)
            red[threadIdx.x] = fmaxf(red[threadIdx.x], red[threadIdx.x + s]);
        __syncthreads();
    }
    if (threadIdx.x == 0) omax[0] = red[0];
}

// ---------------------------------------------------------------------------
// 2) Fold BN into per-channel scale/offset
// ---------------------------------------------------------------------------
__global__ void bnprep_kernel(const float* __restrict__ gamma, const float* __restrict__ beta,
                              const float* __restrict__ mean, const float* __restrict__ var,
                              float* __restrict__ inv, float* __restrict__ bias) {
    int c = threadIdx.x; // 256 threads, 1 block
    float iv = gamma[c] * rsqrtf(var[c] + 1e-5f);
    inv[c]  = iv;
    bias[c] = beta[c] - mean[c] * iv;
}

// ---------------------------------------------------------------------------
// 3) DoReFa weight-quant + pack B in this code's contiguous k order (same as A) for
//    V_WMMA_F32_16X16X32_F16.  Bpack[kt][lane][j]: lane<16 -> N=lane, Kloc=j;
//    lane>=16 -> N=lane-16, Kloc=16+j.  N>=12 zero-padded.
//    K ordering: kt = rs*8 + kc ; k-within-tap c = kc*32 + Kloc ; (r,s)=rs.
// ---------------------------------------------------------------------------
__global__ void packw_kernel(const float* __restrict__ w, const float* __restrict__ maxabs,
                             _Float16* __restrict__ Bpack) {
    int idx = blockIdx.x * 256 + threadIdx.x;       // < KT*512
    if (idx >= KT * 512) return;
    int kt   = idx >> 9;
    int rem  = idx & 511;
    int lane = rem >> 4;
    int j    = rem & 15;
    int n      = lane & 15;
    int klocal = j + ((lane >> 4) << 4);
    int rs = kt >> 3;            // tap index 0..8
    int kc = kt & 7;             // channel chunk 0..7
    int c  = kc * 32 + klocal;   // input channel 0..255
    int r  = rs / 3;
    int s  = rs - r * 3;
    float val = 0.0f;
    if (n < GOUT) {
        float t  = tanhf(w[((n * CIN + c) * 3 + r) * 3 + s]);
        float wn = t / (2.0f * maxabs[0]) + 0.5f;           // in [0,1]
        float q  = rintf(wn * 15.0f) * (1.0f / 15.0f);
        val = 2.0f * q - 1.0f;
    }
    Bpack[idx] = (_Float16)val;
}

// ---------------------------------------------------------------------------
// 4a) Zero the halo border of the NHWC f16 activation tensor [128][34][34][256]
// ---------------------------------------------------------------------------
__global__ void zero_halo_kernel(_Float16* __restrict__ xq) {
    int idx = blockIdx.x * 256 + threadIdx.x;   // over 128*34*34*16 chunks
    if (idx >= BATCH * HP * WP * 16) return;
    int chunk = idx & 15;
    int pix   = idx >> 4;
    int wx    = pix % WP;
    int t     = pix / WP;
    int hy    = t % HP;
    if (hy == 0 || hy == HP - 1 || wx == 0 || wx == WP - 1) {
        uint4 z = {0u, 0u, 0u, 0u};
        *(uint4*)(xq + ((size_t)pix << 8) + (chunk << 4)) = z;
    }
}

// ---------------------------------------------------------------------------
// 4b) Fused BN + act-quant + concat-copy with NCHW -> NHWC(halo) transpose
//     through LDS.  One block = one (b, 32-channel, 32-pixel) tile.
// ---------------------------------------------------------------------------
__global__ void bnquant_kernel(const float* __restrict__ x, const float* __restrict__ inv,
                               const float* __restrict__ bias, float* __restrict__ out,
                               _Float16* __restrict__ xq) {
    __shared__ _Float16 tile[32][33];   // padded stride: no bank conflicts
    const int bid = blockIdx.x;         // 128*8*32 = 32768 blocks
    const int b   = bid >> 8;
    const int rem = bid & 255;
    const int c0  = (rem >> 5) << 5;    // channel tile base (0..224)
    const int hw0 = (rem & 31) << 5;    // pixel tile base (0..992)
    const int tid  = threadIdx.x;
    const int hw_l = tid & 31;
    const int cg4  = tid >> 5;          // 0..7 (wave-uniform)

    // coalesced NCHW reads (lanes sweep hw), concat-copy, quantize into LDS
#pragma unroll
    for (int i = 0; i < 4; ++i) {
        int c_l = cg4 * 4 + i;
        int c   = c0 + c_l;
        size_t gi = (((size_t)b * CIN + c) << 10) + hw0 + hw_l;
        float v = x[gi];
        out[(((size_t)b * COUT_TOTAL + c) << 10) + hw0 + hw_l] = v;   // concat part
        float y = v * inv[c] + bias[c];
        y = fminf(fmaxf(y, 0.0f), 1.0f);
        y = rintf(y * 15.0f) * (1.0f / 15.0f);
        tile[c_l][hw_l] = (_Float16)y;
    }
    __syncthreads();

    // NHWC writes: 8 lanes cover 32 channels of one pixel (64B granular)
    const int hw_l2 = tid >> 3;
    const int cg    = (tid & 7) << 2;
    v4h v;
#pragma unroll
    for (int i = 0; i < 4; ++i) v[i] = tile[cg + i][hw_l2];
    const int hw = hw0 + hw_l2;
    const int h  = hw >> 5;
    const int w  = hw & 31;
    size_t oi = (((size_t)b * HP + (h + 1)) * WP + (w + 1)) * 256 + c0 + cg;
    *(v4h*)(xq + oi) = v;
}

// ---------------------------------------------------------------------------
// 5) Implicit-GEMM 3x3 conv via WMMA f16 -> f32, zero branches in hot loop.
//    One wave: M=16 pixels (half a row), N=16 channels, K=2304 (72 x 32).
//    A: NHWC+halo -> two unconditional global_load_b128 per fragment.
//    B: resident in 72KB dynamic LDS -> two ds_load_b128 per fragment.
// ---------------------------------------------------------------------------
__global__ __launch_bounds__(256, 2)
void wmma_conv_kernel(const _Float16* __restrict__ xq,
                      const _Float16* __restrict__ Bpack,
                      float* __restrict__ out) {
    extern __shared__ _Float16 sB[];    // KT*512 halves = 73728 bytes
    const int tid = threadIdx.x;
    {
        const uint4* g = (const uint4*)Bpack;
        uint4*       l = (uint4*)sB;
        for (int i = tid; i < (KT * 512) / 8; i += 256) l[i] = g[i];
    }
    __syncthreads();

    const int wave   = tid >> 5;
    const int lane   = tid & 31;
    const int laneHi = lane >> 4;
    const int m      = lane & 15;       // pixel row of A this lane holds

    const int b  = blockIdx.x >> 3;                    // 8 blocks per image
    const int h  = ((blockIdx.x & 7) << 2) + (wave >> 1);
    const int w0 = (wave & 1) << 4;

    v8f acc = {0.f, 0.f, 0.f, 0.f, 0.f, 0.f, 0.f, 0.f};

#pragma unroll
    for (int rs = 0; rs < 9; ++rs) {
        const int r = rs / 3;
        const int s = rs - r * 3;
        // halo layout: (h + r - 1) + 1 == h + r ; (w + s - 1) + 1 == w + s
        const _Float16* __restrict__ arow =
            xq + ((((size_t)b * HP + (h + r)) * WP + (w0 + m + s)) << 8);
#pragma unroll
        for (int kc = 0; kc < 8; ++kc) {
            const v8h lo = *(const v8h*)(arow + kc * 32 + laneHi * 16);
            const v8h hi = *(const v8h*)(arow + kc * 32 + laneHi * 16 + 8);
            const v16h a = __builtin_shufflevector(lo, hi,
                0, 1, 2, 3, 4, 5, 6, 7, 8, 9, 10, 11, 12, 13, 14, 15);
            const int kt = rs * 8 + kc;
            const v16h bfrag = *(const v16h*)(sB + kt * 512 + lane * 16);
            acc = __builtin_amdgcn_wmma_f32_16x16x32_f16(
                      false, a, false, bfrag, (short)0, acc, false, false);
        }
    }

    // epilogue: C/D layout -> out[b, 256+n, h, w0 + (laneHi*8 + q)]
    const int n     = lane & 15;
    const int mbase = laneHi * 8;
    if (n < GOUT) {
        float* ob = out + ((size_t)b * COUT_TOTAL + CIN + n) * (HH * WW)
                        + (h << 5) + w0 + mbase;
#pragma unroll
        for (int q = 0; q < 8; ++q) ob[q] = acc[q];
    }
}

// ---------------------------------------------------------------------------
// launch
// ---------------------------------------------------------------------------
extern "C" void kernel_launch(void* const* d_in, const int* in_sizes, int n_in,
                              void* d_out, int out_size, void* d_ws, size_t ws_size,
                              hipStream_t stream) {
    const float* x      = (const float*)d_in[0];
    const float* gamma  = (const float*)d_in[1];
    const float* beta   = (const float*)d_in[2];
    const float* mean   = (const float*)d_in[3];
    const float* var    = (const float*)d_in[4];
    const float* weight = (const float*)d_in[5];
    float* out = (float*)d_out;

    char* ws = (char*)d_ws;
    float*    w_max = (float*)(ws);                 // 1 float
    float*    inv   = (float*)(ws + 64);            // 256 floats
    float*    bias  = (float*)(ws + 64 + 1024);     // 256 floats
    _Float16* Bpack = (_Float16*)(ws + 8192);       // 73728 bytes
    _Float16* xq    = (_Float16*)(ws + 81920);      // [128][34][34][256] f16 ~72.3 MB

    const int nW = GOUT * CIN * 9;

    absmax_kernel<<<dim3(1), dim3(256), 0, stream>>>(weight, w_max, nW);
    bnprep_kernel<<<dim3(1), dim3(256), 0, stream>>>(gamma, beta, mean, var, inv, bias);
    packw_kernel<<<dim3((KT * 512 + 255) / 256), dim3(256), 0, stream>>>(weight, w_max, Bpack);

    const int haloChunks = BATCH * HP * WP * 16;
    zero_halo_kernel<<<dim3((haloChunks + 255) / 256), dim3(256), 0, stream>>>(xq);

    // 128 images * 8 ctiles * 32 hwtiles = 32768 blocks
    bnquant_kernel<<<dim3(32768), dim3(256), 0, stream>>>(x, inv, bias, out, xq);

    // 8192 pixel-tiles, 8 waves per block -> 1024 blocks; 72 KB dynamic LDS
    wmma_conv_kernel<<<dim3(1024), dim3(256), (size_t)(KT * 512 * sizeof(_Float16)), stream>>>(
        xq, Bpack, out);
}
